// VoxelUNet_22222160789826
// MI455X (gfx1250) — matrix operand order repaired and checked by reading, not compiled, not run on hardware
//
#include <hip/hip_runtime.h>

// ---------------------------------------------------------------------------
// MI455X (gfx1250, wave32) sparse-voxel 3-level U-Net.
// Convs are implicit GEMM on v_wmma_f32_16x16x32_bf16. One wave computes a
// 16-voxel x (16*NACC)-outchannel tile: A fragment loaded once per (tap,kc),
// NACC WMMAs issued against NACC packed-B fragments. Boundary handling is
// split per tap: uniform skip (dd/hh OOB), clean interior path (no zero
// movs), per-lane select only on W-edge tiles.
// ---------------------------------------------------------------------------

typedef __bf16 bf16;
typedef __attribute__((ext_vector_type(16))) __bf16 bf16x16;
typedef __attribute__((ext_vector_type(8)))  __bf16 bf16x8;
typedef __attribute__((ext_vector_type(8)))  float  f32x8;

#define NEG_SLOPE 0.2f
#define EPS 1e-5f

__device__ __forceinline__ f32x8 wmma_bf16(bf16x16 a, bf16x16 b, f32x8 c) {
  return __builtin_amdgcn_wmma_f32_16x16x32_bf16(
      /*neg_a=*/false, a, /*neg_b=*/false, b,
      /*c_mod=*/(short)0, c, /*reuse_a=*/false, /*reuse_b=*/false);
}

// A-matrix per-lane fragment: row = voxel (lane&15), 32 K-channels/chunk.
// Lane<16 needs channels {0..7, 16..23}; lane>=16 needs {8..15, 24..31}.
// kb = (lane>>4)*8. Two 16B loads.
__device__ __forceinline__ bf16x16 load_a16(const bf16* __restrict__ row, int kb) {
  bf16x8 lo = *(const bf16x8*)(row + kb);
  bf16x8 hi = *(const bf16x8*)(row + 16 + kb);
  bf16x16 a;
#pragma unroll
  for (int i = 0; i < 8; ++i) { a[i] = lo[i]; a[i + 8] = hi[i]; }
  return a;
}

// ---------------------------------------------------------------------------
// Small utility kernels
// ---------------------------------------------------------------------------

__global__ void k_zero(float* __restrict__ p, int n) {
  int i = blockIdx.x * blockDim.x + threadIdx.x;
  if (i < n) p[i] = 0.f;
}

__global__ void k_mask_init(const unsigned char* __restrict__ mb,
                            float* __restrict__ m0, float* __restrict__ cnt, int V) {
  __shared__ float red[256];
  int t = threadIdx.x;
  int v = blockIdx.x * 256 + t;
  float m = 0.f;
  if (v < V) { m = mb[v] ? 1.f : 0.f; m0[v] = m; }
  red[t] = m; __syncthreads();
  for (int s = 128; s > 0; s >>= 1) { if (t < s) red[t] += red[t + s]; __syncthreads(); }
  if (t == 0) atomicAdd(cnt, red[0]);
}

__global__ void k_mask_pool(const float* __restrict__ mf, float* __restrict__ mc,
                            float* __restrict__ cnt, int Dc, int Hc, int Wc) {
  __shared__ float red[256];
  int t = threadIdx.x;
  int v = blockIdx.x * 256 + t;
  int Vc = Dc * Hc * Wc;
  float m = 0.f;
  if (v < Vc) {
    int w = v % Wc; int q = v / Wc; int h = q % Hc; int d = q / Hc;
    int Hf = Hc * 2, Wf = Wc * 2;
#pragma unroll
    for (int tz = 0; tz < 2; ++tz)
#pragma unroll
      for (int ty = 0; ty < 2; ++ty)
#pragma unroll
        for (int tx = 0; tx < 2; ++tx)
          m = fmaxf(m, mf[(size_t)((2 * d + tz) * Hf + 2 * h + ty) * Wf + 2 * w + tx]);
    mc[v] = m;
  }
  red[t] = m; __syncthreads();
  for (int s = 128; s > 0; s >>= 1) { if (t < s) red[t] += red[t + s]; __syncthreads(); }
  if (t == 0) atomicAdd(cnt, red[0]);
}

// Pack weights [T, Cin, Cout] (f32) into per-lane B fragments (bf16):
// dst[((tap*NT + nt)*KC + kc)*512 + lane*16 + e],
// K = kc*32 + (lane>>4)*8 + (e&7) + (e>>3)*16 ; cout = nt*16 + (lane&15). flip => tap' = T-1-tap.
__global__ void k_pack_w(const float* __restrict__ Wsrc, bf16* __restrict__ Wp,
                         int T, int Cin, int Cout, int NT, int KC, int flip, int total) {
  int idx = blockIdx.x * blockDim.x + threadIdx.x;
  if (idx >= total) return;
  int e = idx & 15;
  int r = idx >> 4;
  int lane = r & 31; r >>= 5;
  int kc = r % KC; r /= KC;
  int nt = r % NT;
  int tap = r / NT;
  int k = kc * 32 + ((lane >> 4) * 8) + (e & 7) + ((e >> 3) * 16);
  int co = nt * 16 + (lane & 15);
  int ts = flip ? (T - 1 - tap) : tap;
  float v = (k < Cin && co < Cout) ? Wsrc[((size_t)ts * Cin + k) * Cout + co] : 0.f;
  Wp[idx] = (bf16)v;
}

// Stem conv: k3 s1 SAME, Cin=4, Cout=32, +bias, *mask -> bf16 packed (stride 32)
__global__ void k_conv_init(const float* __restrict__ X, const float* __restrict__ Wt,
                            const float* __restrict__ bias, const float* __restrict__ m0,
                            bf16* __restrict__ Yb, int D, int H, int W, int total) {
  int idx = blockIdx.x * blockDim.x + threadIdx.x;
  if (idx >= total) return;
  int co = idx & 31;
  int v = idx >> 5;
  int w = v % W; int q = v / W; int h = q % H; int d = q / H;
  float s = bias[co];
  for (int tap = 0; tap < 27; ++tap) {
    int dz = tap / 9 - 1, dy = (tap / 3) % 3 - 1, dx = tap % 3 - 1;
    int dd = d + dz, hh = h + dy, ww = w + dx;
    if ((unsigned)dd < (unsigned)D && (unsigned)hh < (unsigned)H && (unsigned)ww < (unsigned)W) {
      const float* xr = X + (size_t)((dd * H + hh) * W + ww) * 4;
      const float* wr = Wt + (size_t)tap * 4 * 32 + co;
      s += xr[0] * wr[0] + xr[1] * wr[32] + xr[2] * wr[64] + xr[3] * wr[96];
    }
  }
  Yb[(size_t)v * 32 + co] = (bf16)(s * m0[v]);
}

// ---------------------------------------------------------------------------
// WMMA conv kernels. One wave per (16-voxel, 16*NACC-outch) tile.
// ---------------------------------------------------------------------------

// 3x3x3 SAME conv. mask optional. Stores f32 (stride CoutReal) and/or bf16
// packed (stride dstStride, channel offset dstOff).
template <int NACC>
__global__ void k_conv3(const bf16* __restrict__ X, const bf16* __restrict__ Wp,
                        const float* __restrict__ mask,
                        float* __restrict__ Yf, bf16* __restrict__ Yb,
                        int D, int H, int W, int Cin, int CoutReal,
                        int NT, int dstStride, int dstOff, int tiles) {
  int wid = blockIdx.x * (blockDim.x >> 5) + (threadIdx.x >> 5);
  if (wid >= tiles) return;
  int lane = threadIdx.x & 31;
  int NG = NT / NACC;
  int tile = wid / NG;
  int nt0 = (wid % NG) * NACC;
  int vbase = tile * 16;
  int w0 = vbase % W; int q = vbase / W; int h0 = q % H; int d0 = q / H;
  int mrow = lane & 15;
  int kb = (lane >> 4) * 8;
  int KC = Cin >> 5;
  f32x8 acc[NACC];
#pragma unroll
  for (int j = 0; j < NACC; ++j) acc[j] = {};

  for (int tap = 0; tap < 27; ++tap) {
    int dz = tap / 9 - 1, dy = (tap / 3) % 3 - 1, dx = tap % 3 - 1;
    int dd = d0 + dz, hh = h0 + dy;
    if ((unsigned)dd >= (unsigned)D || (unsigned)hh >= (unsigned)H)
      continue;  // whole tap contributes zero (uniform skip)
    int ww = w0 + mrow + dx;
    const bf16* row = X + (size_t)((dd * H + hh) * W + ww) * Cin;
    const bf16* wp = Wp + ((size_t)(tap * NT + nt0) * KC) * 512 + lane * 16;
    bool interior = (w0 + dx >= 0) && (w0 + 15 + dx < W);
    if (interior) {
      for (int kc = 0; kc < KC; ++kc) {
        bf16x16 a = load_a16(row + (size_t)kc * 32, kb);
        const bf16* wpk = wp + (size_t)kc * 512;
#pragma unroll
        for (int j = 0; j < NACC; ++j) {
          bf16x16 b = *(const bf16x16*)(wpk + (size_t)j * KC * 512);
          acc[j] = wmma_bf16(a, b, acc[j]);
        }
      }
    } else {
      bool lv = (unsigned)ww < (unsigned)W;  // only edge lanes differ
      for (int kc = 0; kc < KC; ++kc) {
        bf16x16 a = {};
        if (lv) a = load_a16(row + (size_t)kc * 32, kb);
        const bf16* wpk = wp + (size_t)kc * 512;
#pragma unroll
        for (int j = 0; j < NACC; ++j) {
          bf16x16 b = *(const bf16x16*)(wpk + (size_t)j * KC * 512);
          acc[j] = wmma_bf16(a, b, acc[j]);
        }
      }
    }
  }

  int rbase = (lane >> 4) * 8;
#pragma unroll
  for (int j = 0; j < NACC; ++j) {
    int n = (lane & 15) + (nt0 + j) * 16;
    if (n < CoutReal) {
#pragma unroll
      for (int r = 0; r < 8; ++r) {
        int v = vbase + rbase + r;
        float mv = mask ? mask[v] : 1.f;
        float val = acc[j][r] * mv;
        if (Yf) Yf[(size_t)v * CoutReal + n] = val;
        if (Yb) Yb[(size_t)v * dstStride + dstOff + n] = (bf16)val;
      }
    }
  }
}

// k2 s2 down conv (no padding, no mask per reference). Stores bf16 packed.
template <int NACC>
__global__ void k_down2(const bf16* __restrict__ X, const bf16* __restrict__ Wp,
                        bf16* __restrict__ Yb,
                        int Dc, int Hc, int Wc, int Cin, int CoutReal,
                        int NT, int dstStride, int dstOff, int tiles) {
  int wid = blockIdx.x * (blockDim.x >> 5) + (threadIdx.x >> 5);
  if (wid >= tiles) return;
  int lane = threadIdx.x & 31;
  int NG = NT / NACC;
  int tile = wid / NG;
  int nt0 = (wid % NG) * NACC;
  int vbase = tile * 16;
  int w0 = vbase % Wc; int q = vbase / Wc; int h0 = q % Hc; int d0 = q / Hc;
  int mrow = lane & 15;
  int kb = (lane >> 4) * 8;
  int Hf = Hc * 2, Wf = Wc * 2;
  int KC = Cin >> 5;
  f32x8 acc[NACC];
#pragma unroll
  for (int j = 0; j < NACC; ++j) acc[j] = {};
  for (int tap = 0; tap < 8; ++tap) {
    int tz = tap >> 2, ty = (tap >> 1) & 1, tx = tap & 1;
    int vf = ((2 * d0 + tz) * Hf + 2 * h0 + ty) * Wf + 2 * (w0 + mrow) + tx;
    const bf16* row = X + (size_t)vf * Cin;
    const bf16* wp = Wp + ((size_t)(tap * NT + nt0) * KC) * 512 + lane * 16;
    for (int kc = 0; kc < KC; ++kc) {
      bf16x16 a = load_a16(row + (size_t)kc * 32, kb);
      const bf16* wpk = wp + (size_t)kc * 512;
#pragma unroll
      for (int j = 0; j < NACC; ++j) {
        bf16x16 b = *(const bf16x16*)(wpk + (size_t)j * KC * 512);
        acc[j] = wmma_bf16(a, b, acc[j]);
      }
    }
  }
  int rbase = (lane >> 4) * 8;
#pragma unroll
  for (int j = 0; j < NACC; ++j) {
    int n = (lane & 15) + (nt0 + j) * 16;
    if (n < CoutReal) {
#pragma unroll
      for (int r = 0; r < 8; ++r) {
        int v = vbase + rbase + r;
        Yb[(size_t)v * dstStride + dstOff + n] = (bf16)acc[j][r];
      }
    }
  }
}

// k2 s2 transposed conv: fine(2d+t) uses w[1-t] (flip baked into packed
// weights). A fragments (coarse row) loaded ONCE into registers (KC compile
// time), reused across all 8 taps. Output *= fine mask; bf16 into concat buf.
template <int KC, int NACC>
__global__ void k_up2(const bf16* __restrict__ X, const bf16* __restrict__ Wp,
                      const float* __restrict__ maskF, bf16* __restrict__ Yb,
                      int Dc, int Hc, int Wc, int CoutReal,
                      int NT, int dstStride, int dstOff, int tiles) {
  int wid = blockIdx.x * (blockDim.x >> 5) + (threadIdx.x >> 5);
  if (wid >= tiles) return;
  int lane = threadIdx.x & 31;
  const int Cin = KC * 32;
  int NG = NT / NACC;
  int tile = wid / NG;
  int nt0 = (wid % NG) * NACC;
  int vbase = tile * 16;
  int w0 = vbase % Wc; int q = vbase / Wc; int h0 = q % Hc; int d0 = q / Hc;
  int mrow = lane & 15;
  int kb = (lane >> 4) * 8;
  int rbase = (lane >> 4) * 8;
  int Hf = Hc * 2, Wf = Wc * 2;
  const bf16* rowA = X + (size_t)(vbase + mrow) * Cin;
  bf16x16 a[KC];
#pragma unroll
  for (int kc = 0; kc < KC; ++kc) a[kc] = load_a16(rowA + kc * 32, kb);

  for (int tap = 0; tap < 8; ++tap) {
    int tz = tap >> 2, ty = (tap >> 1) & 1, tx = tap & 1;
    f32x8 acc[NACC];
#pragma unroll
    for (int j = 0; j < NACC; ++j) acc[j] = {};
    const bf16* wp = Wp + ((size_t)(tap * NT + nt0) * KC) * 512 + lane * 16;
#pragma unroll
    for (int kc = 0; kc < KC; ++kc) {
      const bf16* wpk = wp + (size_t)kc * 512;
#pragma unroll
      for (int j = 0; j < NACC; ++j) {
        bf16x16 b = *(const bf16x16*)(wpk + (size_t)j * KC * 512);
        acc[j] = wmma_bf16(a[kc], b, acc[j]);
      }
    }
#pragma unroll
    for (int j = 0; j < NACC; ++j) {
      int n = (lane & 15) + (nt0 + j) * 16;
      if (n < CoutReal) {
#pragma unroll
        for (int r = 0; r < 8; ++r) {
          int wq = w0 + rbase + r;
          int vf = ((2 * d0 + tz) * Hf + (2 * h0 + ty)) * Wf + 2 * wq + tx;
          float val = acc[j][r] * maskF[vf];
          Yb[(size_t)vf * dstStride + dstOff + n] = (bf16)val;
        }
      }
    }
  }
}

// ---------------------------------------------------------------------------
// Instance-norm statistics + activation/pack
// ---------------------------------------------------------------------------

__global__ void k_reduce_stats(const float* __restrict__ X, float* __restrict__ sum,
                               float* __restrict__ sumsq, int V, int C, int VPB) {
  __shared__ float ls[256], lq[256];
  int t = threadIdx.x;
  int c = t % C;
  int sub = t / C;
  int TPC = 256 / C;
  float s = 0.f, qq = 0.f;
  int v0 = blockIdx.x * VPB;
  for (int i = sub; i < VPB; i += TPC) {
    int v = v0 + i;
    if (v < V) {
      float x = X[(size_t)v * C + c];
      s += x; qq += x * x;
    }
  }
  ls[t] = s; lq[t] = qq; __syncthreads();
  for (int st = TPC >> 1; st > 0; st >>= 1) {
    if (sub < st) { ls[t] += ls[t + st * C]; lq[t] += lq[t + st * C]; }
    __syncthreads();
  }
  if (sub == 0) { atomicAdd(&sum[c], ls[t]); atomicAdd(&sumsq[c], lq[t]); }
}

__global__ void k_finalize_stats(const float* __restrict__ sum, const float* __restrict__ sumsq,
                                 const float* __restrict__ counts, int lvl, int C,
                                 float* __restrict__ mean, float* __restrict__ rstd) {
  int c = threadIdx.x;
  if (c >= C) return;
  float n = counts[lvl];
  float mu = sum[c] / n;
  float var = sumsq[c] / n - mu * mu;
  mean[c] = mu;
  rstd[c] = rsqrtf(var + EPS);
}

// mode bit0: instance-norm (needs mean/rstd/mask); bit1: leaky-relu.
__global__ void k_pack_act(const float* __restrict__ X, const float* __restrict__ mask,
                           const float* __restrict__ mean, const float* __restrict__ rstd,
                           float* __restrict__ Yf, bf16* __restrict__ Yb,
                           int V, int C, int dstStride, int dstOff, int mode) {
  int idx = blockIdx.x * blockDim.x + threadIdx.x;
  if (idx >= V * C) return;
  int c = idx % C;
  int v = idx / C;
  float x = X[idx];
  if (mode & 1) x = (x - mean[c]) * rstd[c] * mask[v];
  if (mode & 2) x = x < 0.f ? NEG_SLOPE * x : x;
  if (Yf) Yf[idx] = x;
  Yb[(size_t)v * dstStride + dstOff + c] = (bf16)x;
}

// ---------------------------------------------------------------------------
// Host orchestration
// ---------------------------------------------------------------------------

extern "C" void kernel_launch(void* const* d_in, const int* in_sizes, int n_in,
                              void* d_out, int out_size, void* d_ws, size_t ws_size,
                              hipStream_t stream) {
  (void)in_sizes; (void)n_in; (void)out_size; (void)ws_size;
  const float* x_in    = (const float*)d_in[0];
  const unsigned char* maskb = (const unsigned char*)d_in[1];  // jnp bool -> 1 byte
  const float* w_init  = (const float*)d_in[2];
  const float* b_init  = (const float*)d_in[3];
  const float* db0_w1  = (const float*)d_in[4];
  const float* db0_w2  = (const float*)d_in[5];
  const float* ds0_w   = (const float*)d_in[6];
  const float* db1_w1  = (const float*)d_in[7];
  const float* db1_w2  = (const float*)d_in[8];
  const float* ds1_w   = (const float*)d_in[9];
  const float* db2_w1  = (const float*)d_in[10];
  const float* db2_w2  = (const float*)d_in[11];
  const float* us0_w   = (const float*)d_in[12];
  const float* ub0_w1  = (const float*)d_in[13];
  const float* ub0_w2  = (const float*)d_in[14];
  const float* us1_w   = (const float*)d_in[15];
  const float* ub1_w1  = (const float*)d_in[16];
  const float* ub1_w2  = (const float*)d_in[17];
  const float* out0_w  = (const float*)d_in[18];
  const float* out1_w  = (const float*)d_in[19];
  const float* out2_w  = (const float*)d_in[20];

  const int D0 = 64, H0 = 64, W0 = 64, V0 = D0 * H0 * W0;
  const int D1 = 32, H1 = 32, W1 = 32, V1 = D1 * H1 * W1;
  const int D2 = 16, H2 = 16, W2 = 16, V2 = D2 * H2 * W2;

  // Output layout: o0,o1,o2,f0,f1,f2 concatenated (f32)
  float* o0 = (float*)d_out;
  float* o1 = o0 + (size_t)V2 * 20;
  float* o2 = o1 + (size_t)V1 * 20;
  float* f0 = o2 + (size_t)V0 * 20;
  float* f1 = f0 + (size_t)V2 * 128;
  float* f2 = f1 + (size_t)V1 * 64;

  // ---- workspace bump allocator ----
  char* wsb = (char*)d_ws;
  size_t off = 0;
  auto alloc = [&](size_t bytes) -> void* {
    off = (off + 255) & ~(size_t)255;
    void* p = wsb + off;
    off += bytes;
    return p;
  };
  float* m0 = (float*)alloc((size_t)V0 * 4);
  float* m1 = (float*)alloc((size_t)V1 * 4);
  float* m2 = (float*)alloc((size_t)V2 * 4);
  float* counts = (float*)alloc(16);
  float* ssum = (float*)alloc(256 * 4);   // ssum & ssq contiguous
  float* ssq  = (float*)alloc(256 * 4);
  float* mean = (float*)alloc(256 * 4);
  float* rstd = (float*)alloc(256 * 4);
  bf16* PKA = (bf16*)alloc((size_t)V0 * 32 * 2);
  bf16* PKB = (bf16*)alloc((size_t)V0 * 32 * 2);
  bf16* CC0 = (bf16*)alloc((size_t)V0 * 96 * 2);
  bf16* CC1 = (bf16*)alloc((size_t)V1 * 192 * 2);
  float* CONVR = (float*)alloc((size_t)V0 * 32 * 4);  // also fits V1*64, V2*128

  auto wpk_alloc = [&](int T, int NT, int KC) -> bf16* {
    return (bf16*)alloc((size_t)T * NT * KC * 512 * 2);
  };
  bf16* P_db0w1 = wpk_alloc(27, 2, 1);
  bf16* P_db0w2 = wpk_alloc(27, 2, 1);
  bf16* P_ds0   = wpk_alloc(8, 2, 1);
  bf16* P_db1w1 = wpk_alloc(27, 4, 1);
  bf16* P_db1w2 = wpk_alloc(27, 4, 2);
  bf16* P_ds1   = wpk_alloc(8, 4, 2);
  bf16* P_db2w1 = wpk_alloc(27, 8, 2);
  bf16* P_db2w2 = wpk_alloc(27, 8, 4);
  bf16* P_us0   = wpk_alloc(8, 8, 4);
  bf16* P_ub0w1 = wpk_alloc(27, 4, 6);
  bf16* P_ub0w2 = wpk_alloc(27, 4, 2);
  bf16* P_us1   = wpk_alloc(8, 4, 2);
  bf16* P_ub1w1 = wpk_alloc(27, 2, 3);
  bf16* P_ub1w2 = wpk_alloc(27, 2, 1);
  bf16* P_out0  = wpk_alloc(27, 2, 4);
  bf16* P_out1  = wpk_alloc(27, 2, 2);
  bf16* P_out2  = wpk_alloc(27, 2, 1);

  // ---- launch helpers ----
  auto packw = [&](const float* src, bf16* dst, int T, int Cin, int Cout, int NT,
                   int KC, int flip) {
    int total = T * NT * KC * 512;
    k_pack_w<<<(total + 255) / 256, 256, 0, stream>>>(src, dst, T, Cin, Cout, NT, KC,
                                                      flip, total);
  };
  auto conv3 = [&](const bf16* X, const bf16* Wp, const float* mask, float* Yf,
                   bf16* Yb, int D, int H, int W, int Cin, int CoutReal, int NT,
                   int dstStride, int dstOff) {
    if (NT % 4 == 0) {
      int tiles = (D * H * W / 16) * (NT / 4);
      k_conv3<4><<<(tiles + 3) / 4, 128, 0, stream>>>(X, Wp, mask, Yf, Yb, D, H, W,
                                                      Cin, CoutReal, NT, dstStride,
                                                      dstOff, tiles);
    } else {
      int tiles = (D * H * W / 16) * (NT / 2);
      k_conv3<2><<<(tiles + 3) / 4, 128, 0, stream>>>(X, Wp, mask, Yf, Yb, D, H, W,
                                                      Cin, CoutReal, NT, dstStride,
                                                      dstOff, tiles);
    }
  };
  auto down2 = [&](const bf16* X, const bf16* Wp, bf16* Yb, int Dc, int Hc, int Wc,
                   int Cin, int CoutReal, int NT, int dstStride, int dstOff) {
    if (NT % 4 == 0) {
      int tiles = (Dc * Hc * Wc / 16) * (NT / 4);
      k_down2<4><<<(tiles + 3) / 4, 128, 0, stream>>>(X, Wp, Yb, Dc, Hc, Wc, Cin,
                                                      CoutReal, NT, dstStride, dstOff,
                                                      tiles);
    } else {
      int tiles = (Dc * Hc * Wc / 16) * (NT / 2);
      k_down2<2><<<(tiles + 3) / 4, 128, 0, stream>>>(X, Wp, Yb, Dc, Hc, Wc, Cin,
                                                      CoutReal, NT, dstStride, dstOff,
                                                      tiles);
    }
  };
  auto inorm_pack = [&](const float* convf, const float* mask, int V, int C, int lvl,
                        bf16* dst, int stride) {
    k_zero<<<2, 256, 0, stream>>>(ssum, 512);
    const int VPB = 256;
    k_reduce_stats<<<(V + VPB - 1) / VPB, 256, 0, stream>>>(convf, ssum, ssq, V, C, VPB);
    k_finalize_stats<<<1, 256, 0, stream>>>(ssum, ssq, counts, lvl, C, mean, rstd);
    int total = V * C;
    k_pack_act<<<(total + 255) / 256, 256, 0, stream>>>(convf, mask, mean, rstd,
                                                        nullptr, dst, V, C, stride, 0, 3);
  };
  auto lrelu_pack = [&](const float* convf, int V, int C, float* yf, bf16* dst,
                        int stride, int doff) {
    int total = V * C;
    k_pack_act<<<(total + 255) / 256, 256, 0, stream>>>(convf, nullptr, nullptr, nullptr,
                                                        yf, dst, V, C, stride, doff, 2);
  };

  // ---- masks & counts ----
  k_zero<<<1, 256, 0, stream>>>(counts, 4);
  k_mask_init<<<(V0 + 255) / 256, 256, 0, stream>>>(maskb, m0, &counts[0], V0);
  k_mask_pool<<<(V1 + 255) / 256, 256, 0, stream>>>(m0, m1, &counts[1], D1, H1, W1);
  k_mask_pool<<<(V2 + 255) / 256, 256, 0, stream>>>(m1, m2, &counts[2], D2, H2, W2);

  // ---- weight packing ----
  packw(db0_w1, P_db0w1, 27, 32, 32, 2, 1, 0);
  packw(db0_w2, P_db0w2, 27, 32, 32, 2, 1, 0);
  packw(ds0_w,  P_ds0,   8,  32, 32, 2, 1, 0);
  packw(db1_w1, P_db1w1, 27, 32, 64, 4, 1, 0);
  packw(db1_w2, P_db1w2, 27, 64, 64, 4, 2, 0);
  packw(ds1_w,  P_ds1,   8,  64, 64, 4, 2, 0);
  packw(db2_w1, P_db2w1, 27, 64, 128, 8, 2, 0);
  packw(db2_w2, P_db2w2, 27, 128, 128, 8, 4, 0);
  packw(us0_w,  P_us0,   8, 128, 128, 8, 4, 1);   // transposed conv: tap flip
  packw(ub0_w1, P_ub0w1, 27, 192, 64, 4, 6, 0);
  packw(ub0_w2, P_ub0w2, 27, 64, 64, 4, 2, 0);
  packw(us1_w,  P_us1,   8, 64, 64, 4, 2, 1);     // tap flip
  packw(ub1_w1, P_ub1w1, 27, 96, 32, 2, 3, 0);
  packw(ub1_w2, P_ub1w2, 27, 32, 32, 2, 1, 0);
  packw(out0_w, P_out0,  27, 128, 20, 2, 4, 0);
  packw(out1_w, P_out1,  27, 64, 20, 2, 2, 0);
  packw(out2_w, P_out2,  27, 32, 20, 2, 1, 0);

  // ---- stem ----
  {
    int total = V0 * 32;
    k_conv_init<<<(total + 255) / 256, 256, 0, stream>>>(x_in, w_init, b_init, m0, PKA,
                                                         D0, H0, W0, total);
  }

  // ---- down block 0 (L0, 32ch) ----
  conv3(PKA, P_db0w1, m0, CONVR, nullptr, D0, H0, W0, 32, 32, 2, 0, 0);
  inorm_pack(CONVR, m0, V0, 32, 0, PKB, 32);                 // h
  conv3(PKB, P_db0w2, m0, CONVR, nullptr, D0, H0, W0, 32, 32, 2, 0, 0);
  lrelu_pack(CONVR, V0, 32, nullptr, PKA, 32, 0);            // r0 -> ds0 input
  lrelu_pack(CONVR, V0, 32, nullptr, CC0, 96, 0);            // r0 -> concat(L0) 0..31

  // ---- downsample to L1, down block 1 (64ch) ----
  down2(PKA, P_ds0, PKB, D1, H1, W1, 32, 32, 2, 32, 0);      // x(L1) bf16
  conv3(PKB, P_db1w1, m1, CONVR, nullptr, D1, H1, W1, 32, 64, 4, 0, 0);
  inorm_pack(CONVR, m1, V1, 64, 1, PKA, 64);                 // h
  conv3(PKA, P_db1w2, m1, CONVR, nullptr, D1, H1, W1, 64, 64, 4, 0, 0);
  lrelu_pack(CONVR, V1, 64, nullptr, PKB, 64, 0);            // r1 -> ds1 input
  lrelu_pack(CONVR, V1, 64, nullptr, CC1, 192, 0);           // r1 -> concat(L1) 0..63

  // ---- downsample to L2, bottleneck (128ch) ----
  down2(PKB, P_ds1, PKA, D2, H2, W2, 64, 64, 4, 64, 0);      // x(L2) bf16
  conv3(PKA, P_db2w1, m2, CONVR, nullptr, D2, H2, W2, 64, 128, 8, 0, 0);
  inorm_pack(CONVR, m2, V2, 128, 2, PKB, 128);               // h
  conv3(PKB, P_db2w2, m2, CONVR, nullptr, D2, H2, W2, 128, 128, 8, 0, 0);
  lrelu_pack(CONVR, V2, 128, f0, PKA, 128, 0);               // f0 (f32) + x(L2) bf16

  // ---- head 0 + up to L1 ----
  conv3(PKA, P_out0, m2, o0, nullptr, D2, H2, W2, 128, 20, 2, 0, 0);
  {
    int tiles = (V2 / 16) * (8 / 4);
    k_up2<4, 4><<<(tiles + 3) / 4, 128, 0, stream>>>(PKA, P_us0, m1, CC1, D2, H2, W2,
                                                     128, 8, 192, 64, tiles);
  }

  // ---- up block 0 (192 -> 64) ----
  conv3(CC1, P_ub0w1, m1, CONVR, nullptr, D1, H1, W1, 192, 64, 4, 0, 0);
  inorm_pack(CONVR, m1, V1, 64, 1, PKB, 64);                 // h
  conv3(PKB, P_ub0w2, m1, CONVR, nullptr, D1, H1, W1, 64, 64, 4, 0, 0);
  lrelu_pack(CONVR, V1, 64, f1, PKA, 64, 0);                 // f1 (f32) + x(L1) bf16

  // ---- head 1 + up to L0 ----
  conv3(PKA, P_out1, m1, o1, nullptr, D1, H1, W1, 64, 20, 2, 0, 0);
  {
    int tiles = (V1 / 16) * (4 / 4);
    k_up2<2, 4><<<(tiles + 3) / 4, 128, 0, stream>>>(PKA, P_us1, m0, CC0, D1, H1, W1,
                                                     64, 4, 96, 32, tiles);
  }

  // ---- up block 1 (96 -> 32) ----
  conv3(CC0, P_ub1w1, m0, CONVR, nullptr, D0, H0, W0, 96, 32, 2, 0, 0);
  inorm_pack(CONVR, m0, V0, 32, 0, PKB, 32);                 // h
  conv3(PKB, P_ub1w2, m0, CONVR, nullptr, D0, H0, W0, 32, 32, 2, 0, 0);
  lrelu_pack(CONVR, V0, 32, f2, PKA, 32, 0);                 // f2 (f32) + x(L0) bf16

  // ---- head 2 ----
  conv3(PKA, P_out2, m0, o2, nullptr, D0, H0, W0, 32, 20, 2, 0, 0);
}
